// GraphRes_46969762349780
// MI455X (gfx1250) — compile-verified, matrix operand order checked by reading, not run on hardware
//
#include <hip/hip_runtime.h>
#include <hip/hip_bf16.h>

typedef __attribute__((ext_vector_type(16))) _Float16 v16h;
typedef __attribute__((ext_vector_type(8)))  float    v8f;

#define TPB 256

__device__ __forceinline__ bool finitef_(float v) {
    return (__float_as_uint(v) & 0x7f800000u) != 0x7f800000u;
}

// Order-independent (deterministic) float atomic max.
// Positive floats compare like ints; negative floats compare inversely as uints.
// -0 is canonicalized to +0 so the sign split is consistent.
__device__ __forceinline__ void atomicMaxFloat(float* addr, float val) {
    float v = val + 0.0f;
    if (v >= 0.0f) atomicMax((int*)addr, __float_as_int(v));
    else           atomicMin((unsigned int*)addr, __float_as_uint(v));
}

__global__ void fill_u32_k(unsigned int* __restrict__ p, unsigned int val, int n) {
    int i = blockIdx.x * TPB + threadIdx.x;
    if (i < n) p[i] = val;
}

// ---------------- Layer 1 (cin=1, cout=16): trivial K=3 GEMM on VALU --------
// S[i,c] = x[i]*W[0,c] + px*W[1,c] + py*W[2,c]
__global__ void layer1_s_k(const float* __restrict__ x, const float* __restrict__ pos,
                           const float* __restrict__ W, float* __restrict__ S, int N) {
    int gid = blockIdx.x * TPB + threadIdx.x;
    int i = gid >> 4, c = gid & 15;
    if (i >= N) return;
    float s = x[i]*W[c] + pos[2*i]*W[16+c] + pos[2*i+1]*W[32+c];
    S[(size_t)i*16 + c] = s;
}

// ---- Pack W (row-major [cin+2,32], x-rows only, K padded to 32) into the ----
// ---- CDNA5 wave32 B-operand layout for V_WMMA_F32_16X16X32_F16.         ----
// B is 32(K)x16(N) per tile; lane<16 -> col=lane, K={0..7,16..23};
// lane>=16 -> col=lane-16, K={8..15,24..31}; 2 halves per dword in K order.
__global__ void prep_wb_k(const float* __restrict__ W, int cinReal,
                          _Float16* __restrict__ WB) {
    int t = blockIdx.x * TPB + threadIdx.x;
    if (t >= 1024) return;               // 2 tiles * 32 lanes * 16 halves
    int tile = t >> 9;
    int lane = (t >> 4) & 31;
    int j    = t & 15;
    int k = ((j < 8) ? j : (j + 8)) + ((lane < 16) ? 0 : 8);
    int n = tile*16 + (lane & 15);
    float v = (k < cinReal) ? W[k*32 + n] : 0.0f;   // zero-pad K>=cinReal
    WB[t] = (_Float16)v;
}

// ---------------- WMMA node GEMM: S[N,32] = H(f16)[N,32] @ Wx + pos @ Wp ----
// One wave per 16-node tile. A operand per the ISA 16-bit A 16x32 layout:
// lane<16 holds row m=lane, K {0..7,16..23}; lane>=16 holds K {8..15,24..31}.
__global__ void gemm_wmma_k(const _Float16* __restrict__ H, const _Float16* __restrict__ WB,
                            const float* __restrict__ pos, const float* __restrict__ W,
                            int cin, float* __restrict__ S, int nwaves) {
    int gtid = blockIdx.x * TPB + threadIdx.x;
    int wv = gtid >> 5;
    if (wv >= nwaves) return;            // wave-uniform guard: EXEC stays all-1s
    int lane = gtid & 31;
    int nodeBase = wv << 4;
    int m  = lane & 15;
    int hi = lane >> 4;
    int k0 = hi << 3;                    // 0 or 8
    union { v16h v; float4 f4[2]; } A, B0, B1;
    const _Float16* hrow = H + ((size_t)(nodeBase + m) << 5);
    A.f4[0] = *(const float4*)(hrow + k0);        // K = k0..k0+7
    A.f4[1] = *(const float4*)(hrow + k0 + 16);   // K = k0+16..k0+23
    const _Float16* wb = WB + lane * 16;
    B0.f4[0] = *(const float4*)(wb);
    B0.f4[1] = *(const float4*)(wb + 8);
    B1.f4[0] = *(const float4*)(wb + 512);
    B1.f4[1] = *(const float4*)(wb + 520);
    v8f z = {};
    union { v8f v; float f[8]; } C0, C1;
    C0.v = __builtin_amdgcn_wmma_f32_16x16x32_f16(false, A.v, false, B0.v,
                                                  (short)0, z, false, false);
    C1.v = __builtin_amdgcn_wmma_f32_16x16x32_f16(false, A.v, false, B1.v,
                                                  (short)0, z, false, false);
    // Epilogue: add K=2 pos term, store. C/D layout: lane<16 -> M=r, N=lane;
    // lane>=16 -> M=8+r, N=lane-16.
    int chA = m, chB = m + 16;
    float wp0a = W[cin*32 + chA], wp1a = W[(cin+1)*32 + chA];
    float wp0b = W[cin*32 + chB], wp1b = W[(cin+1)*32 + chB];
    int nb = nodeBase + (hi << 3);
    #pragma unroll
    for (int r = 0; r < 8; ++r) {
        int node = nb + r;
        float px = pos[2*node], py = pos[2*node+1];
        float* srow = S + ((size_t)node << 5);
        srow[chA] = C0.f[r] + px*wp0a + py*wp1a;
        srow[chB] = C1.f[r] + px*wp0b + py*wp1b;
    }
}

// ---------------- Edge scatter-max: M[dst,c] = max(M[dst,c], S[src,c]) ------
// One wave per edge (C=32): lanes = channels, src/dst wave-uniform,
// 128B coalesced gather of S[src] out of L2.
__global__ void scatter_max_k(const float* __restrict__ S, const int* __restrict__ src,
                              const int* __restrict__ dst, float* __restrict__ M,
                              int clog2, int total) {
    int gid = blockIdx.x * TPB + threadIdx.x;
    if (gid >= total) return;
    int e = gid >> clog2;
    int c = gid & ((1 << clog2) - 1);
    int s = src[e], d = dst[e];
    float v = S[((size_t)s << clog2) + c];
    atomicMaxFloat(&M[((size_t)d << clog2) + c], v);
}

// ---------------- BN stats: agg = finite(M) ? M - pos@Wp : 0 ----------------
// Fixed 512-block grid-stride, fixed LDS tree -> deterministic partials.
__global__ void stats_k(const float* __restrict__ M, const float* __restrict__ pos,
                        const float* __restrict__ W, int cin, int clog2,
                        float* __restrict__ partial, int N) {
    int C = 1 << clog2;
    int t = threadIdx.x;
    int c = t & (C - 1);
    int rl = t >> clog2;
    int rpb = TPB >> clog2;
    float wp0 = W[cin*C + c], wp1 = W[(cin+1)*C + c];
    float s = 0.f, ss = 0.f;
    for (int i = blockIdx.x * rpb + rl; i < N; i += gridDim.x * rpb) {
        float m = M[((size_t)i << clog2) + c];
        float a = 0.f;
        if (finitef_(m)) a = m - (pos[2*i]*wp0 + pos[2*i+1]*wp1);
        s += a; ss += a*a;
    }
    __shared__ float sm[TPB], sq[TPB];
    sm[t] = s; sq[t] = ss;
    __syncthreads();
    for (int str = TPB/2; str >= C; str >>= 1) {
        if (t < str) { sm[t] += sm[t+str]; sq[t] += sq[t+str]; }
        __syncthreads();
    }
    if (t < C) {
        partial[blockIdx.x*(2*C) + t]     = sm[t];
        partial[blockIdx.x*(2*C) + C + t] = sq[t];
    }
}

// Fixed-order final reduction -> ab[c] = gamma*invstd, ab[C+c] = beta - mean*that
__global__ void bnfinal_k(const float* __restrict__ partial, const float* __restrict__ gamma,
                          const float* __restrict__ beta, int C, int nblk, float invN,
                          float* __restrict__ ab) {
    __shared__ float tot[64];
    int t = threadIdx.x;                 // 64 threads
    float s = 0.f;
    int stride = 2*C;
    if (t < stride) for (int b = 0; b < nblk; ++b) s += partial[b*stride + t];
    tot[t] = s;
    __syncthreads();
    if (t < C) {
        float mean = tot[t]*invN;
        float var  = tot[C+t]*invN - mean*mean;
        float a = gamma[t] * rsqrtf(var + 1e-5f);
        ab[t] = a;
        ab[C+t] = beta[t] - mean*a;
    }
}

// Normalize + ReLU. Writes f16 [N,32] (zero-padded for C=16) for the next
// WMMA layer, or f32 for the final pooling layer.
__global__ void normalize_k(const float* __restrict__ M, const float* __restrict__ pos,
                            const float* __restrict__ W, const float* __restrict__ ab,
                            int cin, int clog2, int writeF32,
                            _Float16* __restrict__ Hout, float* __restrict__ Fout, int N) {
    int gid = blockIdx.x * TPB + threadIdx.x;
    int i = gid >> 5, c = gid & 31;
    if (i >= N) return;
    int C = 1 << clog2;
    float y = 0.f;
    if (c < C) {
        float m = M[((size_t)i << clog2) + c];
        float a = 0.f;
        if (finitef_(m)) a = m - (pos[2*i]*W[cin*C + c] + pos[2*i+1]*W[(cin+1)*C + c]);
        y = fmaxf(a*ab[c] + ab[C + c], 0.f);
    }
    if (writeF32) Fout[(size_t)i*32 + c] = y;
    else          Hout[(size_t)i*32 + c] = (_Float16)y;
}

// ---------------- Grid max-pool: seg = batch*64 + cellx*8 + celly ----------
__global__ void pool_k(const float* __restrict__ H4, const float* __restrict__ pos,
                       const int* __restrict__ batch, float* __restrict__ pooled, int N) {
    int gid = blockIdx.x * TPB + threadIdx.x;
    int i = gid >> 5, c = gid & 31;
    if (i >= N) return;
    float px = pos[2*i], py = pos[2*i+1];
    int cx = (int)floorf(px / 15.0f);  cx = min(max(cx, 0), 7);
    int cy = (int)floorf(py / 12.5f);  cy = min(max(cy, 0), 7);
    int seg = batch[i]*64 + cx*8 + cy;
    atomicMaxFloat(&pooled[(size_t)seg*32 + c], H4[(size_t)i*32 + c]);
}

// out[b,o] = sum_j fin(pooled[b*2048+j]) * fcW[o*2048+j]   (flat layout matches)
__global__ void fc_k(const float* __restrict__ pooled, const float* __restrict__ fcW,
                     float* __restrict__ out) {
    int blk = blockIdx.x;                // 32 blocks = (b,o) pairs
    int b = blk >> 1, o = blk & 1;
    int t = threadIdx.x;
    float acc = 0.f;
    for (int j = t; j < 2048; j += TPB) {
        float v = pooled[b*2048 + j];
        if (!finitef_(v)) v = 0.f;       // empty cells -> 0
        acc += v * fcW[o*2048 + j];
    }
    __shared__ float sm[TPB];
    sm[t] = acc;
    __syncthreads();
    for (int s = TPB/2; s > 0; s >>= 1) {
        if (t < s) sm[t] += sm[t+s];
        __syncthreads();
    }
    if (t == 0) out[b*2 + o] = sm[0];
}

extern "C" void kernel_launch(void* const* d_in, const int* in_sizes, int n_in,
                              void* d_out, int out_size, void* d_ws, size_t ws_size,
                              hipStream_t stream) {
    (void)n_in; (void)out_size; (void)ws_size;
    const int N = in_sizes[0];           // 500000 (divisible by 16)
    const int E = in_sizes[2] / 2;       // 4000000
    const float* x     = (const float*)d_in[0];
    const float* pos   = (const float*)d_in[1];
    const int*   ei    = (const int*)d_in[2];
    const int*   batch = (const int*)d_in[3];
    const float* W1 = (const float*)d_in[4];
    const float* g1 = (const float*)d_in[5];
    const float* b1 = (const float*)d_in[6];
    const float* W2 = (const float*)d_in[7];
    const float* g2 = (const float*)d_in[8];
    const float* b2 = (const float*)d_in[9];
    const float* W3 = (const float*)d_in[10];
    const float* g3 = (const float*)d_in[11];
    const float* b3 = (const float*)d_in[12];
    const float* W4 = (const float*)d_in[13];
    const float* g4 = (const float*)d_in[14];
    const float* b4 = (const float*)d_in[15];
    const float* fcW= (const float*)d_in[16];
    float* out = (float*)d_out;
    const int* srcp = ei;
    const int* dstp = ei + E;

    // Workspace layout (~160.3 MB total), all 16B-aligned:
    char* ws = (char*)d_ws;
    float*    S   = (float*)ws;                               // N*32 f32  (reused as f32 H4)
    float*    M   = (float*)(ws + (size_t)N*32*4);            // N*32 f32
    _Float16* Hf  = (_Float16*)(ws + (size_t)N*32*8);         // N*32 f16
    _Float16* WB  = (_Float16*)(ws + (size_t)N*32*10);        // 3*1024 f16 B operands
    float* partial= (float*)(ws + (size_t)N*32*10 + 8192);    // 512*64 f32
    float* ab     = partial + 512*64;                         // 64 f32
    float* pooled = ab + 64;                                  // 1024*32 f32

    const int STATS_BLOCKS = 512;
    const float invN = 1.0f / (float)N;

    // Pack weight B-operands once (layers 2..4; layer 2 K-padded 16->32).
    prep_wb_k<<<4, TPB, 0, stream>>>(W2, 16, WB);
    prep_wb_k<<<4, TPB, 0, stream>>>(W3, 32, WB + 1024);
    prep_wb_k<<<4, TPB, 0, stream>>>(W4, 32, WB + 2048);

    auto layer = [&](int cin, int clog2, const float* W, const float* g, const float* bt,
                     const _Float16* wb, int writeF32) {
        int C = 1 << clog2;
        int nMC = N * C;
        fill_u32_k<<<(nMC + TPB - 1)/TPB, TPB, 0, stream>>>((unsigned int*)M, 0xFF800000u, nMC);
        if (wb == nullptr) {
            layer1_s_k<<<(N*16 + TPB - 1)/TPB, TPB, 0, stream>>>(x, pos, W, S, N);
        } else {
            int nwaves = N / 16;
            gemm_wmma_k<<<(nwaves*32 + TPB - 1)/TPB, TPB, 0, stream>>>(Hf, wb, pos, W,
                                                                       cin, S, nwaves);
        }
        int total = E << clog2;
        scatter_max_k<<<(total + TPB - 1)/TPB, TPB, 0, stream>>>(S, srcp, dstp, M, clog2, total);
        stats_k<<<STATS_BLOCKS, TPB, 0, stream>>>(M, pos, W, cin, clog2, partial, N);
        bnfinal_k<<<1, 64, 0, stream>>>(partial, g, bt, C, STATS_BLOCKS, invN, ab);
        normalize_k<<<(N*32 + TPB - 1)/TPB, TPB, 0, stream>>>(M, pos, W, ab, cin, clog2,
                                                              writeF32, Hf, S, N);
    };

    layer( 1, 4, W1, g1, b1, nullptr,   0);   // 1 -> 16, VALU GEMM
    layer(16, 5, W2, g2, b2, WB,        0);   // 16 -> 32, WMMA (K zero-padded)
    layer(32, 5, W3, g3, b3, WB + 1024, 0);   // 32 -> 32, WMMA
    layer(32, 5, W4, g4, b4, WB + 2048, 1);   // 32 -> 32, WMMA; f32 out into S

    fill_u32_k<<<(1024*32 + TPB - 1)/TPB, TPB, 0, stream>>>((unsigned int*)pooled,
                                                            0xFF800000u, 1024*32);
    pool_k<<<(N*32 + TPB - 1)/TPB, TPB, 0, stream>>>(S, pos, batch, pooled, N);
    fc_k<<<32, TPB, 0, stream>>>(pooled, fcW, out);
}
